// MetaGAT_47974784696685
// MI455X (gfx1250) — matrix operand order repaired and checked by reading, not compiled, not run on hardware
//
#include <hip/hip_runtime.h>
#include <math.h>

typedef __attribute__((ext_vector_type(16))) _Float16 v16h;
typedef __attribute__((ext_vector_type(8)))  float    v8f;

#define N_   500
#define DEG_ 16
#define B_   8
#define T_   12
#define H_   64
#define F_   32
#define BT_  96
#define M1_  16

// ---------------- LDS layout (bytes) ----------------
static constexpr int W3FRAG_HALFS = 4 * 4 * 32 * 16;   // [ht][chunk][lane][16] = 8192 halves
static constexpr int ASRC_HALFS   = 16 * 2 * 32 * 16;  // [edge][chunk<2][lane][16] = 16384
static constexpr int ADST_HALFS   = 2 * 32 * 16;       // [chunk-2][lane][16] = 1024

static constexpr int OFF_W3A  = 0;
static constexpr int OFF_W3B  = OFF_W3A  + W3FRAG_HALFS * 2;   // 16384
static constexpr int OFF_B3   = OFF_W3B  + W3FRAG_HALFS * 2;   // 32768
static constexpr int OFF_ASRC = OFF_B3   + W3FRAG_HALFS * 2;   // 49152
static constexpr int OFF_ADST = OFF_ASRC + ASRC_HALFS   * 2;   // 81920
static constexpr int OFF_H1   = OFF_ADST + ADST_HALFS   * 2;   // 83968  (16*16 f32)
static constexpr int OFF_COEF = OFF_H1   + 16 * 16 * 4;        // 84992  (16*2 f32)
static constexpr int OFF_SRCE = OFF_COEF + 16 * 2 * 4;         // 85120  (16 int)
static constexpr int OFF_REDA = OFF_SRCE + 64;                 // 85184  (8*32*8 f32)
static constexpr int OFF_REDD = OFF_REDA + 8 * 32 * 8 * 4;     // 93376
static constexpr int OFF_REDN = OFF_REDD + 8 * 32 * 8 * 4;     // 101568
static constexpr int SMEM_BYTES = OFF_REDN + 8 * 32 * 8 * 4;   // 109760

__device__ __forceinline__ float sigmoidf_(float x) { return 1.0f / (1.0f + __expf(-x)); }

// A-fragment K mapping for 16-bit 16x32 A tiles (ISA 7.12.2):
// lane 0-15:  V0..3 -> K 0..7,  V4..7 -> K 16..23
// lane 16-31: V0..3 -> K 8..15, V4..7 -> K 24..31
__device__ __forceinline__ int a_kloc(int lane, int i) {
    int v = i >> 1;
    return ((v >> 2) << 4) + (((lane >> 4) & 1) << 3) + ((v & 3) << 1) + (i & 1);
}
// B-fragment K mapping for 16-bit 32x16 B tiles (same as A):
// lane 0-15: K 0..7 then 16..23; lane 16-31: K 8..15 then 24..31
__device__ __forceinline__ int b_kloc(int lane, int i) {
    return ((i >> 3) << 4) + (((lane >> 4) & 1) << 3) + (i & 7);
}

__global__ __launch_bounds__(256)
void MetaGAT_kernel(const float* __restrict__ state, const float* __restrict__ feature,
                    const float* __restrict__ dist,  const float* __restrict__ W1,
                    const float* __restrict__ b1,    const float* __restrict__ W2,
                    const float* __restrict__ b2,    const float* __restrict__ W3,
                    const float* __restrict__ b3,    const float* __restrict__ gate,
                    const int*   __restrict__ src,   float* __restrict__ out)
{
    extern __shared__ char smem[];
    _Float16* w3aF  = (_Float16*)(smem + OFF_W3A);
    _Float16* w3bF  = (_Float16*)(smem + OFF_W3B);
    _Float16* b3F   = (_Float16*)(smem + OFF_B3);
    _Float16* asrcF = (_Float16*)(smem + OFF_ASRC);
    _Float16* adstF = (_Float16*)(smem + OFF_ADST);
    float*    h1s   = (float*)(smem + OFF_H1);
    float*    coef  = (float*)(smem + OFF_COEF);
    int*      srcE  = (int*)(smem + OFF_SRCE);
    float*    redA  = (float*)(smem + OFF_REDA);
    float*    redD  = (float*)(smem + OFF_REDD);
    float*    redN  = (float*)(smem + OFF_REDN);

    const int n    = blockIdx.y;       // dst node
    const int bt0  = blockIdx.x << 4;  // bt-tile base row
    const int tid  = threadIdx.x;
    const int lane = tid & 31;
    const int wv   = tid >> 5;

    // Warm L2 with W3 (64 KB): one cacheline per thread -> global_prefetch_b8
    __builtin_prefetch(W3 + tid * 64, 0, 1);

    // ---- Phase 1a: W3/b3 B-fragments (f16, swizzled) -------------------------
    for (int f = tid; f < W3FRAG_HALFS; f += 256) {
        int i  = f & 15;
        int ln = (f >> 4) & 31;
        int ch = (f >> 9) & 3;
        int ht = f >> 11;
        int k  = (ch << 5) + b_kloc(ln, i);       // 0..127 (row of per-edge w)
        int h  = (ht << 4) + (ln & 15);           // 0..63
        w3aF[f] = (_Float16)W3[k * H_ + h];
        w3bF[f] = (_Float16)W3[2 * H_ * H_ + k * H_ + h];
        b3F[f]  = (_Float16)b3[k * H_ + h];
    }
    // ---- Phase 1b: shared dst-half A-fragments (K chunks 2,3 of st) ----------
    for (int f = tid; f < ADST_HALFS; f += 256) {
        int i  = f & 15;
        int ln = (f >> 4) & 31;
        int c2 = f >> 9;                           // 0,1 -> global chunks 2,3
        int row = ln & 15;
        int k   = (c2 << 5) + a_kloc(ln, i);       // 0..63 = h-index of dst state
        adstF[f] = (_Float16)state[((bt0 + row) * N_ + n) * H_ + k];
    }
    // ---- Phase 1c: hypernet layer 1 (one thread per (edge, hidden)) ----------
    {
        int e = tid >> 4, m = tid & 15;
        int eidx = n * DEG_ + e;
        int sn = src[eidx];
        float acc = b1[m];
        #pragma unroll 4
        for (int d = 0; d < F_; ++d)
            acc += feature[sn * F_ + d] * W1[d * M1_ + m];
        #pragma unroll 4
        for (int d = 0; d < F_; ++d)
            acc += feature[n * F_ + d] * W1[(F_ + d) * M1_ + m];
        acc += dist[eidx] * W1[2 * F_ * M1_ + m];
        h1s[e * 16 + m] = sigmoidf_(acc);
        if (m == 0) srcE[e] = sn;
    }
    __syncthreads();

    // ---- Phase 2a: hypernet layer 2 -> per-edge coefficients -----------------
    if (tid < 32) {
        int e = tid >> 1, p = tid & 1;
        float acc = b2[p];
        #pragma unroll
        for (int m = 0; m < M1_; ++m)
            acc += h1s[e * 16 + m] * W2[m * 2 + p];
        coef[e * 2 + p] = sigmoidf_(acc);
    }
    // ---- Phase 2b: per-edge src-half A-fragments (K chunks 0,1) --------------
    for (int f = tid; f < ASRC_HALFS; f += 256) {
        int i  = f & 15;
        int ln = (f >> 4) & 31;
        int ch = (f >> 9) & 1;
        int e  = f >> 10;
        int row = ln & 15;
        int k   = (ch << 5) + a_kloc(ln, i);       // 0..63 = h-index of src state
        int sn  = srcE[e];
        asrcF[f] = (_Float16)state[((bt0 + row) * N_ + sn) * H_ + k];
    }
    __syncthreads();

    // ---- Phase 3: 16x16x32 f16 WMMA GEMMs: alpha[e, bt-tile, h-tile] ---------
    const int ht = wv & 3;          // h-tile (16 cols)
    const int eg = wv >> 2;         // edge octet (0: e0-7, 1: e8-15)

    v8f acc[8];
    #pragma unroll
    for (int e8 = 0; e8 < 8; ++e8)
        #pragma unroll
        for (int r = 0; r < 8; ++r) acc[e8][r] = 0.0f;

    _Float16 c0[8], c1[8];
    #pragma unroll
    for (int e8 = 0; e8 < 8; ++e8) {
        c0[e8] = (_Float16)coef[(eg * 8 + e8) * 2 + 0];
        c1[e8] = (_Float16)coef[(eg * 8 + e8) * 2 + 1];
    }

    #pragma unroll
    for (int ch = 0; ch < 4; ++ch) {
        const v16h wa = *(const v16h*)&w3aF[(((ht << 2) + ch) * 32 + lane) * 16];
        const v16h wb = *(const v16h*)&w3bF[(((ht << 2) + ch) * 32 + lane) * 16];
        const v16h bb = *(const v16h*)&b3F [(((ht << 2) + ch) * 32 + lane) * 16];
        v16h ad;
        #pragma unroll
        for (int i = 0; i < 16; ++i) ad[i] = (_Float16)0.0f;
        if (ch >= 2) ad = *(const v16h*)&adstF[(((ch - 2) * 32) + lane) * 16];

        #pragma unroll
        for (int e8 = 0; e8 < 8; ++e8) {
            v16h af = (ch >= 2)
                ? ad
                : *(const v16h*)&asrcF[((((eg * 8 + e8) * 2 + ch) * 32) + lane) * 16];
            v16h bf;
            #pragma unroll
            for (int i = 0; i < 16; ++i)
                bf[i] = (_Float16)(c0[e8] * wa[i] + c1[e8] * wb[i] + bb[i]);
            acc[e8] = __builtin_amdgcn_wmma_f32_16x16x32_f16(
                false, af, false, bf, (short)0, acc[e8], false, false);
        }
    }

    // ---- Phase 4: LeakyReLU + segment softmax over 16 edges + aggregation ---
    #pragma unroll
    for (int e8 = 0; e8 < 8; ++e8)
        #pragma unroll
        for (int r = 0; r < 8; ++r) {
            float v = acc[e8][r];
            acc[e8][r] = v > 0.0f ? v : 0.01f * v;
        }

    float mx[8];
    #pragma unroll
    for (int r = 0; r < 8; ++r) {
        float m = acc[0][r];
        #pragma unroll
        for (int e8 = 1; e8 < 8; ++e8) m = fmaxf(m, acc[e8][r]);
        mx[r] = m;
    }
    #pragma unroll
    for (int r = 0; r < 8; ++r) redA[(wv * 32 + lane) * 8 + r] = mx[r];
    __syncthreads();
    const int partner = ((eg ^ 1) << 2) + ht;
    #pragma unroll
    for (int r = 0; r < 8; ++r)
        mx[r] = fmaxf(mx[r], redA[(partner * 32 + lane) * 8 + r]);

    const int hcol = (ht << 4) + (lane & 15);
    const int rbase = bt0 + ((lane >> 4) << 3);

    float den[8], num[8];
    #pragma unroll
    for (int r = 0; r < 8; ++r) { den[r] = 0.0f; num[r] = 0.0f; }
    #pragma unroll
    for (int e8 = 0; e8 < 8; ++e8) {
        int sn = srcE[eg * 8 + e8];
        #pragma unroll
        for (int r = 0; r < 8; ++r) {
            float p  = __expf(acc[e8][r] - mx[r]);
            float sv = state[((rbase + r) * N_ + sn) * H_ + hcol];  // src_s gather (L2)
            den[r] += p;
            num[r] += p * sv;
        }
    }
    #pragma unroll
    for (int r = 0; r < 8; ++r) {
        redD[(wv * 32 + lane) * 8 + r] = den[r];
        redN[(wv * 32 + lane) * 8 + r] = num[r];
    }
    __syncthreads();

    if (eg == 0) {
        const float gs = sigmoidf_(gate[0]);
        #pragma unroll
        for (int r = 0; r < 8; ++r) {
            float D  = den[r] + redD[(partner * 32 + lane) * 8 + r];
            float Nu = num[r] + redN[(partner * 32 + lane) * 8 + r];
            float o  = Nu / D;
            o = (o > 0.0f ? o : 0.0f) * gs;
            out[((rbase + r) * N_ + n) * H_ + hcol] = o;   // [B,T,N,H]
        }
    }
}

extern "C" void kernel_launch(void* const* d_in, const int* in_sizes, int n_in,
                              void* d_out, int out_size, void* d_ws, size_t ws_size,
                              hipStream_t stream) {
    (void)in_sizes; (void)n_in; (void)out_size; (void)d_ws; (void)ws_size;
    const float* state   = (const float*)d_in[0];
    const float* feature = (const float*)d_in[1];
    const float* dist    = (const float*)d_in[2];
    const float* W1      = (const float*)d_in[3];
    const float* b1      = (const float*)d_in[4];
    const float* W2      = (const float*)d_in[5];
    const float* b2      = (const float*)d_in[6];
    const float* W3      = (const float*)d_in[7];
    const float* b3      = (const float*)d_in[8];
    const float* gate    = (const float*)d_in[9];
    const int*   src     = (const int*)d_in[10];
    float*       out     = (float*)d_out;

    static bool attr_done = false;
    if (!attr_done) {
        hipFuncSetAttribute((const void*)MetaGAT_kernel,
                            hipFuncAttributeMaxDynamicSharedMemorySize, SMEM_BYTES);
        attr_done = true;
    }

    dim3 grid(BT_ / 16, N_);   // 6 bt-tiles x 500 dst nodes
    dim3 block(256);           // 8 wave32: 4 h-tiles x 2 edge-octets
    MetaGAT_kernel<<<grid, block, SMEM_BYTES, stream>>>(
        state, feature, dist, W1, b1, W2, b2, W3, b3, gate, src, out);
}
